// MyEGAT_35485019799704
// MI455X (gfx1250) — matrix operand order repaired and checked by reading, not compiled, not run on hardware
//
#include <hip/hip_runtime.h>
#include <hip/hip_fp16.h>
#include <math.h>

typedef __attribute__((ext_vector_type(16))) _Float16 v16h;
typedef __attribute__((ext_vector_type(8)))  _Float16 v8h;
typedef __attribute__((ext_vector_type(8)))  float    v8f;
typedef __attribute__((ext_vector_type(4)))  int      v4i;

static constexpr int H  = 4;
static constexpr int OE = 32;
static constexpr int ON = 64;
static constexpr int HN = H * OE;   // 128
static constexpr int HO = H * ON;   // 256
static constexpr int TB = 256;

#if defined(__gfx1250__) && __has_builtin(__builtin_amdgcn_global_load_async_to_lds_b128) && \
    __has_builtin(__builtin_amdgcn_s_wait_asynccnt)
#define USE_ASYNC_LDS 1
typedef __attribute__((address_space(1))) v4i* gv4i_ptr;
typedef __attribute__((address_space(3))) v4i* lv4i_ptr;
#else
#define USE_ASYNC_LDS 0
#endif

// ---------------------------------------------------------------- embed
__global__ void k_embed(const int* __restrict__ types, const float* __restrict__ table,
                        float* __restrict__ nf, int n) {
    int idx = blockIdx.x * blockDim.x + threadIdx.x;
    if (idx >= n * 128) return;
    int i = idx >> 7, c = idx & 127;
    nf[idx] = table[types[i] * 128 + c];
}

// ---------------------------------------------------------------- f32 -> f16
__global__ void k_cvt_h(const float* __restrict__ in, _Float16* __restrict__ out, int count) {
    int idx = blockIdx.x * blockDim.x + threadIdx.x;
    if (idx < count) out[idx] = (_Float16)in[idx];
}

// ---------------------------------------------------------------- fill
__global__ void k_fill(float* __restrict__ p, float v, int count) {
    int idx = blockIdx.x * blockDim.x + threadIdx.x;
    if (idx < count) p[idx] = v;
}

// ---------------------------------------------------------------- WMMA GEMM
// C(R x Ncols) = A(R x K) * B(K x Ncols) [+ bias], f16 in, f32 accumulate.
// One wave per 16x16 C tile, WPB waves/block (tiles along M).
//  - A row-slab (WPB*16 rows x K) staged row-major in LDS; on gfx1250 via the
//    async global->LDS b128 DMA path (ASYNCcnt), layout-preserving.
//  - B slab (K x 16 for this block's column group) staged transposed
//    (column-major, bank-padded) with plain stores.
// All inner-loop operand reads are 128-bit LDS loads.
// A-operand layout (ISA 7.12.2, 16x32 f16): lane g*16+l holds row l's halves
//   a[0..7]  = A[row][kk + g*8 .. +7]      (two contiguous 16B chunks)
//   a[8..15] = A[row][kk + 16 + g*8 .. +7]
// B-operand layout (32x16 f16): lane g*16+l holds col l, same K order as A (b[0..7] <-> kk+g*8.., b[8..15] <-> kk+16+g*8..).
template<int K, int WPB>
__global__ void k_gemm_wmma(const _Float16* __restrict__ A, const _Float16* __restrict__ B,
                            const float* __restrict__ bias, float* __restrict__ C,
                            int R, int Ncols) {
    constexpr int NT   = WPB * 32;    // threads per block
    constexpr int RPB  = WPB * 16;    // rows per block
    constexpr int CPR  = K / 8;       // 16B chunks per A row
    constexpr int KP   = K + 8;       // padded B column stride (halves)

    const int tid  = threadIdx.x;
    const int lane = tid & 31;
    const int wave = tid >> 5;
    const int row0 = blockIdx.x * RPB;
    const int col_base = blockIdx.y * 16;
    const int g = lane >> 4;
    const int l = lane & 15;

    int rows = R - row0; if (rows > RPB) rows = RPB;   // multiple of 16

    __shared__ _Float16 Al[RPB * K];
    __shared__ _Float16 Bl[16 * KP];

#if USE_ASYNC_LDS
    auto* ldsA = (__attribute__((address_space(3))) _Float16*)Al;
    for (int t = tid; t < rows * CPR; t += NT) {       // 16B chunks: (row, c)
        int row = t / CPR, c = t % CPR;
        __builtin_amdgcn_global_load_async_to_lds_b128(
            (gv4i_ptr)(A + (size_t)(row0 + row) * K + c * 8),
            (lv4i_ptr)(ldsA + row * K + c * 8), 0, 0);
    }
#else
    for (int t = tid; t < rows * CPR; t += NT) {
        int row = t / CPR, c = t % CPR;
        *(v8h*)&Al[row * K + c * 8] = *(const v8h*)(A + (size_t)(row0 + row) * K + c * 8);
    }
#endif
    for (int t = tid; t < K * 16; t += NT) {           // B transpose staging
        int k = t >> 4, n = t & 15;
        Bl[n * KP + k] = B[(size_t)k * Ncols + col_base + n];
    }
#if USE_ASYNC_LDS
    __builtin_amdgcn_s_wait_asynccnt(0);
#endif
    __syncthreads();

    const int row_base = wave * 16;                    // within block
    if (row_base < rows) {                             // wave-uniform: EXEC all-ones at WMMA
        const _Float16* Ap = &Al[(row_base + l) * K];
        const _Float16* Bp = &Bl[l * KP];
        v8f c = {};
        #pragma unroll
        for (int kk = 0; kk < K; kk += 32) {
            v8h a0 = *(const v8h*)(Ap + kk + g * 8);
            v8h a1 = *(const v8h*)(Ap + kk + 16 + g * 8);
            v16h a = __builtin_shufflevector(a0, a1, 0, 1, 2, 3, 4, 5, 6, 7,
                                             8, 9, 10, 11, 12, 13, 14, 15);
            v8h b0 = *(const v8h*)(Bp + kk + g * 8);
            v8h b1 = *(const v8h*)(Bp + kk + 16 + g * 8);
            v16h b = __builtin_shufflevector(b0, b1, 0, 1, 2, 3, 4, 5, 6, 7,
                                             8, 9, 10, 11, 12, 13, 14, 15);
            c = __builtin_amdgcn_wmma_f32_16x16x32_f16(false, a, false, b, (short)0, c,
                                                       false, false);
        }
        const float badd = bias ? bias[col_base + l] : 0.0f;
        #pragma unroll
        for (int r = 0; r < 8; ++r) {                  // D layout: VGPR r <-> M = g*8 + r
            int row = row0 + row_base + g * 8 + r;
            C[(size_t)row * Ncols + col_base + l] = c[r] + badd;
        }
    }
}

// ---------------------------------------------------------------- edge combine + logits
// One wave per (edge, head): lane = channel within head (oe == 32 == wave32).
// ef_out may alias ffij (in-place): each element read-then-written by one lane.
__global__ void k_edge_combine(const int* __restrict__ src, const int* __restrict__ dst,
                               const float* __restrict__ fni, const float* __restrict__ fnj,
                               const float* __restrict__ ffij, const float* __restrict__ bias,
                               const float* __restrict__ attn,
                               float* __restrict__ ef_out, float* __restrict__ logits, int E) {
    const int lane = threadIdx.x & 31;
    const int wave = threadIdx.x >> 5;
    const long eh = (long)blockIdx.x * 8 + wave;
    if (eh >= (long)E * H) return;             // wave-uniform
    const int e = (int)(eh >> 2);
    const int h = (int)(eh & 3);
    const int cidx = h * OE + lane;
    const int s = src[e], d = dst[e];
    float x = fni[(size_t)s * HN + cidx] + fnj[(size_t)d * HN + cidx] +
              ffij[(size_t)e * HN + cidx] + bias[cidx];
    float f = (x > 0.f) ? x : 0.01f * x;       // leaky_relu(0.01)
    float p = f * attn[cidx];
    #pragma unroll
    for (int m = 16; m > 0; m >>= 1) p += __shfl_xor(p, m, 32);
    if (lane == 0) logits[(size_t)e * H + h] = p;
    ef_out[(size_t)e * HN + cidx] = (f > 0.f) ? f : (expf(f) - 1.f);  // elu
}

// ---------------------------------------------------------------- float atomic max
__device__ inline void atomicMaxFloat(float* addr, float v) {
    if (v >= 0.f) atomicMax((int*)addr, __float_as_int(v));
    else          atomicMin((unsigned int*)addr, __float_as_uint(v));
}

__global__ void k_seg_max(const int* __restrict__ dst, const float* __restrict__ logits,
                          float* __restrict__ m, int E) {
    int idx = blockIdx.x * blockDim.x + threadIdx.x;
    if (idx >= E * H) return;
    int e = idx >> 2, h = idx & 3;
    atomicMaxFloat(&m[dst[e] * H + h], logits[idx]);
}

__global__ void k_exp_sum(const int* __restrict__ dst, float* __restrict__ logits,
                          const float* __restrict__ m, float* __restrict__ s, int E) {
    int idx = blockIdx.x * blockDim.x + threadIdx.x;
    if (idx >= E * H) return;
    int e = idx >> 2, h = idx & 3;
    float v = expf(logits[idx] - m[dst[e] * H + h]);
    logits[idx] = v;
    atomicAdd(&s[dst[e] * H + h], v);
}

// ---------------------------------------------------------------- weighted scatter-add
__global__ void k_message(const int* __restrict__ src, const int* __restrict__ dst,
                          const float* __restrict__ logits, const float* __restrict__ s,
                          const float* __restrict__ hsrc, float* __restrict__ hacc, int E) {
    long idx = (long)blockIdx.x * blockDim.x + threadIdx.x;
    if (idx >= (long)E * HO) return;
    int e = (int)(idx >> 8);
    int c = (int)(idx & 255);
    int h = c >> 6;                            // head = c / on
    int d = dst[e];
    float alpha = logits[(size_t)e * H + h] / s[(size_t)d * H + h];
    atomicAdd(&hacc[(size_t)d * HO + c], hsrc[(size_t)src[e] * HO + c] * alpha);
}

// ---------------------------------------------------------------- elu
__global__ void k_elu(const float* __restrict__ in, float* __restrict__ out, int count) {
    int idx = blockIdx.x * blockDim.x + threadIdx.x;
    if (idx >= count) return;
    float x = in[idx];
    out[idx] = (x > 0.f) ? x : (expf(x) - 1.f);
}

// ================================================================ host
static inline int cdiv(long a, long b) { return (int)((a + b - 1) / b); }

static void launch_gemm(const _Float16* A, const _Float16* B, const float* bias, float* C,
                        int R, int K, int Ncols, hipStream_t stream) {
    switch (K) {
        case 64:
            k_gemm_wmma<64, 8><<<dim3(cdiv(R, 128), Ncols / 16), 256, 0, stream>>>(
                A, B, bias, C, R, Ncols);
            break;
        case 128:
            k_gemm_wmma<128, 8><<<dim3(cdiv(R, 128), Ncols / 16), 256, 0, stream>>>(
                A, B, bias, C, R, Ncols);
            break;
        case 256:   // 4 waves/block keeps the A slab at 32KB of LDS
            k_gemm_wmma<256, 4><<<dim3(cdiv(R, 64), Ncols / 16), 128, 0, stream>>>(
                A, B, bias, C, R, Ncols);
            break;
        default: break;  // all call sites are 64/128/256
    }
}

extern "C" void kernel_launch(void* const* d_in, const int* in_sizes, int n_in,
                              void* d_out, int out_size, void* d_ws, size_t ws_size,
                              hipStream_t stream) {
    const int*   node_types = (const int*)  d_in[0];
    const int*   src        = (const int*)  d_in[1];
    const int*   dst        = (const int*)  d_in[2];
    const float* efeats     = (const float*)d_in[3];
    const float* embed      = (const float*)d_in[4];
    const float* Wsrc[2] = { (const float*)d_in[5],  (const float*)d_in[12] };
    const float* bsrc[2] = { (const float*)d_in[6],  (const float*)d_in[13] };
    const float* Wni [2] = { (const float*)d_in[7],  (const float*)d_in[14] };
    const float* Wnj [2] = { (const float*)d_in[8],  (const float*)d_in[15] };
    const float* Wfij[2] = { (const float*)d_in[9],  (const float*)d_in[16] };
    const float* attn[2] = { (const float*)d_in[10], (const float*)d_in[17] };
    const float* bias[2] = { (const float*)d_in[11], (const float*)d_in[18] };
    const int in_n[2] = { 128, 256 };
    const int in_e[2] = { 64, 128 };

    const int Nn = in_sizes[0];   // 50000
    const int Ee = in_sizes[1];   // 500000

    // ---- workspace carve-out
    char* ws = (char*)d_ws;
    size_t off = 0;
    auto alloc = [&](size_t bytes) -> void* {
        void* p = ws + off;
        off += (bytes + 255) & ~(size_t)255;
        return p;
    };
    float*    nfA    = (float*)   alloc((size_t)Nn * HO * 4);  // layer input node feats
    float*    nfB    = (float*)   alloc((size_t)Nn * HO * 4);  // h_new accumulator
    _Float16* nf_h   = (_Float16*)alloc((size_t)Nn * HO * 2);
    float*    fni    = (float*)   alloc((size_t)Nn * HN * 4);
    float*    fnj    = (float*)   alloc((size_t)Nn * HN * 4);
    float*    hsrc   = (float*)   alloc((size_t)Nn * HO * 4);
    float*    ffij   = (float*)   alloc((size_t)Ee * HN * 4);  // also carries ef between layers
    _Float16* ef_h   = (_Float16*)alloc((size_t)Ee * HN * 2);
    float*    logits = (float*)   alloc((size_t)Ee * H  * 4);
    float*    mbuf   = (float*)   alloc((size_t)Nn * H  * 4);
    float*    sbuf   = (float*)   alloc((size_t)Nn * H  * 4);
    _Float16* wh0    = (_Float16*)alloc((size_t)256 * 256 * 2);
    _Float16* wh1    = (_Float16*)alloc((size_t)256 * 256 * 2);
    _Float16* wh2    = (_Float16*)alloc((size_t)256 * 256 * 2);
    _Float16* wh3    = (_Float16*)alloc((size_t)256 * 256 * 2);
    (void)ws_size; (void)n_in; (void)out_size;

    float* out_nf = (float*)d_out;                  // (N, 256)
    float* out_ef = out_nf + (size_t)Nn * HO;       // (E, 128)

    // ---- layer 0 node features from embedding table
    k_embed<<<cdiv((long)Nn * 128, TB), TB, 0, stream>>>(node_types, embed, nfA, Nn);

    for (int L = 0; L < 2; ++L) {
        const int kn = in_n[L], ke = in_e[L];
        const float* ef_in  = (L == 0) ? efeats : ffij;
        float*       ef_out = (L == 0) ? ffij   : out_ef;  // L0: in-place over ffij
        float*       nf_out = (L == 0) ? nfA    : out_nf;

        // 1) convert activations + weights to f16 (ef cvt precedes the ffij GEMM overwrite)
        k_cvt_h<<<cdiv((long)Nn * kn, TB), TB, 0, stream>>>(nfA,     nf_h, Nn * kn);
        k_cvt_h<<<cdiv((long)Ee * ke, TB), TB, 0, stream>>>(ef_in,   ef_h, Ee * ke);
        k_cvt_h<<<cdiv((long)kn * HN, TB), TB, 0, stream>>>(Wni[L],  wh0,  kn * HN);
        k_cvt_h<<<cdiv((long)kn * HN, TB), TB, 0, stream>>>(Wnj[L],  wh1,  kn * HN);
        k_cvt_h<<<cdiv((long)ke * HN, TB), TB, 0, stream>>>(Wfij[L], wh2,  ke * HN);
        k_cvt_h<<<cdiv((long)kn * HO, TB), TB, 0, stream>>>(Wsrc[L], wh3,  kn * HO);

        // 2) WMMA GEMMs
        launch_gemm(nf_h, wh0, nullptr, fni,  Nn, kn, HN, stream);
        launch_gemm(nf_h, wh1, nullptr, fnj,  Nn, kn, HN, stream);
        launch_gemm(nf_h, wh3, bsrc[L], hsrc, Nn, kn, HO, stream);
        launch_gemm(ef_h, wh2, nullptr, ffij, Ee, ke, HN, stream);

        // 3) per-edge combine: leaky-relu features, attention logits, elu edge output
        k_edge_combine<<<cdiv((long)Ee * H, 8), TB, 0, stream>>>(
            src, dst, fni, fnj, ffij, bias[L], attn[L], ef_out, logits, Ee);

        // 4) edge softmax over incoming edges + weighted aggregation
        k_fill<<<cdiv((long)Nn * H, TB),  TB, 0, stream>>>(mbuf, -INFINITY, Nn * H);
        k_fill<<<cdiv((long)Nn * H, TB),  TB, 0, stream>>>(sbuf, 0.0f,      Nn * H);
        k_fill<<<cdiv((long)Nn * HO, TB), TB, 0, stream>>>(nfB,  0.0f,      Nn * HO);
        k_seg_max<<<cdiv((long)Ee * H, TB), TB, 0, stream>>>(dst, logits, mbuf, Ee);
        k_exp_sum<<<cdiv((long)Ee * H, TB), TB, 0, stream>>>(dst, logits, mbuf, sbuf, Ee);
        k_message<<<cdiv((long)Ee * HO, TB), TB, 0, stream>>>(src, dst, logits, sbuf, hsrc, nfB, Ee);

        // 5) node activation (layer0 -> nfA feeds layer1; layer1 -> d_out)
        k_elu<<<cdiv((long)Nn * HO, TB), TB, 0, stream>>>(nfB, nf_out, Nn * HO);
    }
}
